// MYLSTM_1_15831249453422
// MI455X (gfx1250) — matrix operand order repaired and checked by reading, not compiled, not run on hardware
//
#include <hip/hip_runtime.h>
#include <hip/hip_bf16.h>
#include <cstdint>

// ---------------- problem constants (match reference) ----------------
#define TSTEPS 49
#define UNITS  256
#define DIMX   256
#define CATK   512          // concat dim = UNITS + DIMX
#define QN     9
#define BATCH  2048
#define ROWS   32           // batch rows per block (two WMMA M-tiles: B-frag reuse x2)
#define NWAVES 16           // 512 threads; wave w owns units [16w, 16w+16)
#define CAT_STRIDE 520      // 512 + 8 halfs pad; 1040B row stride (16B aligned)
#define S_STRIDE   264      // 256 + 8 halfs pad; 528B row stride (16B aligned)

typedef __attribute__((ext_vector_type(16))) __bf16       bf16x16;
typedef __attribute__((ext_vector_type(8)))  float        f32x8;
typedef __attribute__((ext_vector_type(4)))  unsigned int u32x4;
typedef __attribute__((ext_vector_type(4)))  float        f32x4;

union FragB { bf16x16 v; u32x4 u[2]; };

__device__ __forceinline__ unsigned short f2bf(float f) {
  union { float f; unsigned int u; } c; c.f = f;
  unsigned int u = c.u + 0x7FFFu + ((c.u >> 16) & 1u);   // round-to-nearest-even
  return (unsigned short)(u >> 16);
}
__device__ __forceinline__ float sigmoid_fast(float x) { return 1.0f / (1.0f + __expf(-x)); }
__device__ __forceinline__ float tanh_fast(float x) {
  float e = __expf(-2.0f * x);
  return (1.0f - e) / (1.0f + e);
}
__device__ __forceinline__ f32x8 zero8() {
  f32x8 z;
  #pragma unroll
  for (int i = 0; i < 8; ++i) z[i] = 0.0f;
  return z;
}

// ---------------------------------------------------------------------
// Prep kernel 1: pack [Wf|Wi|Wc|Wo] (each 512x256 f32, row-major K x N)
// into bf16 WMMA B-fragments.  Fragment (g, ks, ut): lane l holds
// N = ut*16 + l%16; half s (16 contiguous) has K = ks*32 + 8*(l/16) + s%8 + 16*(s/8), per the
// CDNA5 B-matrix layout (lanes enumerate N; halfs enumerate K).
// FW byte offset = ((g*16 + ks)*16 + ut)*1024 + lane*32.  Total 1 MB.
// ---------------------------------------------------------------------
__global__ __launch_bounds__(256) void pack_gate_weights(
    const float* __restrict__ Wf, const float* __restrict__ Wi,
    const float* __restrict__ Wc, const float* __restrict__ Wo,
    unsigned short* __restrict__ FW) {
  int idx = blockIdx.x * 256 + threadIdx.x;     // < 4*16*16*32*16 = 524288
  int fr   = idx >> 9;
  int rem  = idx & 511;
  int lane = rem >> 4;
  int e    = rem & 15;
  int g  = fr >> 8;
  int ks = (fr >> 4) & 15;
  int ut = fr & 15;
  int K = ks * 32 + ((e & 8) << 1) + 8 * (lane >> 4) + (e & 7);
  int N = ut * 16 + (lane & 15);
  const float* W = (g == 0) ? Wf : (g == 1) ? Wi : (g == 2) ? Wc : Wo;
  FW[idx] = f2bf(W[(size_t)K * UNITS + N]);
}

// ---------------------------------------------------------------------
// Prep kernel 2: U[t][i][j] = sum_q Q[i][j*9+q] * fourier(t)[q], packed as
// bf16 B-fragments per timestep: frag (t, ks, jt): lane l holds
// N(=j) = jt*16 + l%16; half s: K(=i) = ks*32 + 8*(l/16) + s%8 + 16*(s/8).  49 * 128 KB = 6.4 MB.
// ---------------------------------------------------------------------
__global__ __launch_bounds__(256) void pack_proj(
    const float* __restrict__ Q, unsigned short* __restrict__ FU) {
  int idx = blockIdx.x * 256 + threadIdx.x;     // < 49 * 65536
  int t   = idx >> 16;
  int r2  = idx & 65535;
  int fr  = r2 >> 9;
  int rem = r2 & 511;
  int lane = rem >> 4;
  int e    = rem & 15;
  int ks = fr >> 4;
  int jt = fr & 15;
  int i = ks * 32 + ((e & 8) << 1) + 8 * (lane >> 4) + (e & 7);
  int j = jt * 16 + (lane & 15);
  float tt = (float)t / (float)(TSTEPS - 1);
  float bas[QN];
  bas[0] = 1.0f;
  const float SQ2 = 1.41421356237309515f;
  #pragma unroll
  for (int m = 1; m <= (QN - 1) / 2; ++m) {
    float w = 6.283185307179586477f * (float)m * tt;
    bas[2 * m - 1] = SQ2 * __sinf(w);
    bas[2 * m]     = SQ2 * __cosf(w);
  }
  const float* q = Q + (size_t)i * (UNITS * QN) + (size_t)j * QN;
  float acc = 0.0f;
  #pragma unroll
  for (int qq = 0; qq < QN; ++qq) acc = fmaf(q[qq], bas[qq], acc);
  FU[idx] = f2bf(acc);
}

// ---------------------------------------------------------------------
// Main recurrent kernel.  64 blocks x 32 batch rows x 16 waves.
// Wave w owns units [16w, 16w+16): all four gates for those units land in
// identical C-fragment slots, so the c-state update is pure per-lane VALU.
// Each B fragment feeds two M-tiles (2 WMMAs per 1KB L2 fetch).
// All B addressing = uniform saddr + 32-bit lane offset + imm offsets.
// ---------------------------------------------------------------------
__global__ __launch_bounds__(512, 1) void lstm_wmma(
    const float* __restrict__ x, const unsigned short* __restrict__ FW,
    const unsigned short* __restrict__ FU, float* __restrict__ out) {
  __shared__ unsigned short cat[ROWS * CAT_STRIDE];   // bf16 [c | x_t], 33.3 KB
  __shared__ unsigned short sb [ROWS * S_STRIDE];     // bf16 s,         16.9 KB

  const int tid  = threadIdx.x;
  const int wave = tid >> 5;
  const int lane = tid & 31;
  const int hi   = lane >> 4;     // half-wave select
  const int ln   = lane & 15;
  const int row0 = blockIdx.x * ROWS;

  // c-state in C/D fragment layout: (VGPR r, lane) -> row mt*16 + r + 8*hi,
  // unit 16*wave + ln,  mt = M-tile 0/1.
  f32x8 cst[2];
  cst[0] = zero8(); cst[1] = zero8();

  // per-thread x staging: 32 rows x 256 cols f32 -> bf16, 16 floats each
  const int xrow = tid >> 4;            // 0..31
  const int xcol = (tid & 15) * 16;     // 0..240
  const float* xbase = x + ((size_t)(row0 + xrow) * TSTEPS) * DIMX + xcol;

  // divergent 32-bit byte offsets into the fragment buffers
  const unsigned vbase = (unsigned)wave * 1024u + (unsigned)lane * 32u;
  const char* fw8 = (const char*)FW;

  for (int t = 0; t < TSTEPS; ++t) {
    // --- stage cat = [bf16(c) | bf16(x_t)] in LDS ---
    #pragma unroll
    for (int mt = 0; mt < 2; ++mt) {
      const int u = 16 * wave + ln;
      #pragma unroll
      for (int r = 0; r < 8; ++r)
        cat[(mt * 16 + r + 8 * hi) * CAT_STRIDE + u] = f2bf(cst[mt][r]);
    }
    {
      const f32x4* xp = reinterpret_cast<const f32x4*>(xbase + (size_t)t * DIMX);
      unsigned short* cp = &cat[xrow * CAT_STRIDE + UNITS + xcol];
      #pragma unroll
      for (int qv = 0; qv < 4; ++qv) {
        f32x4 v = xp[qv];
        #pragma unroll
        for (int c4 = 0; c4 < 4; ++c4) cp[qv * 4 + c4] = f2bf(v[c4]);
      }
      if (t + 1 < TSTEPS)                                   // next-step x tile
        __builtin_prefetch(xbase + (size_t)(t + 1) * DIMX, 0, 1);
    }
    __syncthreads();

    // --- gate GEMM: (32 x 512) @ (512 x 4*256); wave w: its 16 units x 4 gates x 2 M-tiles ---
    f32x8 acc[4][2];                        // [gate][m-tile]
    #pragma unroll
    for (int g = 0; g < 4; ++g) { acc[g][0] = zero8(); acc[g][1] = zero8(); }

    unsigned kb = 0;                        // ks * 16384 bytes
    #pragma unroll 2
    for (int ks = 0; ks < 16; ++ks) {
      // A fragments (16x32 bf16), ISA layout: halfs 0-7: K=ks*32+hi*8.., halfs 8-15: +16
      FragB a0, a1;
      {
        const unsigned short* ap0 = &cat[ln * CAT_STRIDE + ks * 32 + hi * 8];
        const unsigned short* ap1 = ap0 + 16 * CAT_STRIDE;
        a0.u[0] = *reinterpret_cast<const u32x4*>(ap0);
        a0.u[1] = *reinterpret_cast<const u32x4*>(ap0 + 16);
        a1.u[0] = *reinterpret_cast<const u32x4*>(ap1);
        a1.u[1] = *reinterpret_cast<const u32x4*>(ap1 + 16);
      }
      #pragma unroll
      for (int g = 0; g < 4; ++g) {
        const char* p = fw8 + (vbase + kb + (unsigned)(g * 262144));
        FragB b;
        b.u[0] = *reinterpret_cast<const u32x4*>(p);
        b.u[1] = *reinterpret_cast<const u32x4*>(p + 16);
        acc[g][0] = __builtin_amdgcn_wmma_f32_16x16x32_bf16(
            false, a0.v, false, b.v, (short)0, acc[g][0], false, false);
        acc[g][1] = __builtin_amdgcn_wmma_f32_16x16x32_bf16(
            false, a1.v, false, b.v, (short)0, acc[g][1], false, false);
      }
      kb += 16384u;
    }

    // --- elementwise LSTM combine, all in registers (fragment layout) ---
    #pragma unroll
    for (int mt = 0; mt < 2; ++mt) {
      const int u = 16 * wave + ln;
      #pragma unroll
      for (int r = 0; r < 8; ++r) {
        float fg = sigmoid_fast(acc[0][mt][r]);
        float ig = sigmoid_fast(acc[1][mt][r]);
        float gg = sigmoid_fast(acc[2][mt][r]);   // sigmoid, per reference
        float og = tanh_fast(acc[3][mt][r]);      // tanh, per reference
        float cn = ig * gg + fg * cst[mt][r];
        cst[mt][r] = cn;
        float s = og * tanh_fast(cn);
        sb[(mt * 16 + r + 8 * hi) * S_STRIDE + u] = f2bf(s);
      }
    }
    __syncthreads();

    // --- time-varying projection: h = tanh(s @ U_t), (32x256)@(256x256) ---
    f32x8 hacc[2];
    hacc[0] = zero8(); hacc[1] = zero8();
    const char* fu8 = (const char*)FU + (size_t)t * 131072;   // uniform per-step base
    unsigned ku = 0;
    #pragma unroll
    for (int ks = 0; ks < 8; ++ks) {
      FragB a0, a1;
      {
        const unsigned short* ap0 = &sb[ln * S_STRIDE + ks * 32 + hi * 8];
        const unsigned short* ap1 = ap0 + 16 * S_STRIDE;
        a0.u[0] = *reinterpret_cast<const u32x4*>(ap0);
        a0.u[1] = *reinterpret_cast<const u32x4*>(ap0 + 16);
        a1.u[0] = *reinterpret_cast<const u32x4*>(ap1);
        a1.u[1] = *reinterpret_cast<const u32x4*>(ap1 + 16);
      }
      const char* p = fu8 + (vbase + ku);
      FragB b;
      b.u[0] = *reinterpret_cast<const u32x4*>(p);
      b.u[1] = *reinterpret_cast<const u32x4*>(p + 16);
      hacc[0] = __builtin_amdgcn_wmma_f32_16x16x32_bf16(
          false, a0.v, false, b.v, (short)0, hacc[0], false, false);
      hacc[1] = __builtin_amdgcn_wmma_f32_16x16x32_bf16(
          false, a1.v, false, b.v, (short)0, hacc[1], false, false);
      ku += 16384u;
    }

    // --- write h_t (f32) straight from C-fragments ---
    #pragma unroll
    for (int mt = 0; mt < 2; ++mt) {
      const int j = wave * 16 + ln;
      #pragma unroll
      for (int r = 0; r < 8; ++r) {
        const int m = row0 + mt * 16 + r + 8 * hi;
        out[((size_t)m * TSTEPS + t) * UNITS + j] = tanh_fast(hacc[mt][r]);
      }
    }
    __syncthreads();   // protect cat/sb before next iteration rewrites them
  }
}

// ---------------------------------------------------------------------
extern "C" void kernel_launch(void* const* d_in, const int* in_sizes, int n_in,
                              void* d_out, int out_size, void* d_ws, size_t ws_size,
                              hipStream_t stream) {
  const float* x  = (const float*)d_in[0];
  const float* Wf = (const float*)d_in[1];
  const float* Wi = (const float*)d_in[2];
  const float* Wc = (const float*)d_in[3];
  const float* Wo = (const float*)d_in[4];
  const float* Q  = (const float*)d_in[5];

  unsigned short* FW = (unsigned short*)d_ws;                       // 1 MB
  unsigned short* FU = (unsigned short*)((char*)d_ws + (1 << 20));  // 6.4 MB

  pack_gate_weights<<<2048, 256, 0, stream>>>(Wf, Wi, Wc, Wo, FW);
  pack_proj<<<TSTEPS * 256, 256, 0, stream>>>(Q, FU);
  lstm_wmma<<<BATCH / ROWS, 512, 0, stream>>>(x, FW, FU, (float*)d_out);
}
